// TimeMixingModule_4234837754303
// MI455X (gfx1250) — compile-verified, matrix operand order checked by reading, not run on hardware
//
#include <hip/hip_runtime.h>
#include <hip/hip_bf16.h>

// ---------------------------------------------------------------------------
// RWKV time-mixing for MI455X (gfx1250, wave32, WMMA bf16, async LDS staging)
//   N=8, T=2048, C=1024, NUM_CLS=1024, M = N*T = 16384
// ---------------------------------------------------------------------------

#define NB    8
#define TLEN  2048
#define CDIM  1024
#define MROWS (NB * TLEN)   // 16384
#define NCLS  1024

#define BM 128
#define BN 128
#define KC 32               // K-chunk == WMMA K for bf16
#define LDK 40              // ushort stride: 80B rows -> 16B aligned for async
                            // b128 writes; 20-dword stride => conflict-free
                            // 16-row fragment reads (gcd(20,64)=4, 20i%64 distinct)

typedef __attribute__((ext_vector_type(16))) __bf16 v16bf;
typedef __attribute__((ext_vector_type(8)))  float  v8f;
typedef int v4i __attribute__((vector_size(4 * sizeof(int))));

union FragU { v16bf v; unsigned int u[8]; };

// Async global->LDS path (CDNA5 GLOBAL_LOAD_ASYNC_TO_LDS_B128, ASYNCcnt),
// guarded so the file compiles on toolchains without the builtins.
#if defined(__HIP_DEVICE_COMPILE__) && \
    __has_builtin(__builtin_amdgcn_global_load_async_to_lds_b128) && \
    __has_builtin(__builtin_amdgcn_s_wait_asynccnt)
#define USE_ASYNC_LDS 1
#else
#define USE_ASYNC_LDS 0
#endif

#define GLOBAL_AS __attribute__((address_space(1)))
#define LDS_AS    __attribute__((address_space(3)))

static __device__ __forceinline__ void async_copy16(const void* g, void* l) {
#if USE_ASYNC_LDS
    __builtin_amdgcn_global_load_async_to_lds_b128(
        (GLOBAL_AS v4i*)g, (LDS_AS v4i*)l, 0, 0);
#else
    (void)g; (void)l;
#endif
}

static __device__ __forceinline__ unsigned short f32_to_bf16(float f) {
    union { float f; unsigned int u; } c; c.f = f;
    unsigned int r = c.u + 0x7FFFu + ((c.u >> 16) & 1u);  // RNE
    return (unsigned short)(r >> 16);
}
static __device__ __forceinline__ float bf16_to_f32(unsigned short h) {
    union { unsigned int u; float f; } c; c.u = ((unsigned int)h) << 16;
    return c.f;
}

// Per-lane K-pair dword index inside a bf16 16x32 A/B fragment (ISA 7.12.2):
//   VGPR pr (0..7) holds bf16 pair at K = (pr>=4?16:0) + 8*half + (pr&3)*2
static __device__ __forceinline__ int frag_kpair(int pr, int half) {
    return ((pr >= 4) ? 8 : 0) + 4 * half + (pr & 3);
}

// Stage a 128x32 bf16 tile from a bf16 global matrix (ld = CDIM) into LDS.
// Raw copy: async global->LDS when available, else b128 via VGPRs.
static __device__ __forceinline__ void stage_tile_bf16(
    const unsigned short* __restrict__ G, size_t row0, int k0,
    unsigned short (*S)[LDK], int tid)
{
    #pragma unroll
    for (int it = 0; it < 2; ++it) {
        int idx = tid + it * 256;        // over 128 rows x 4 (16B segments)
        int row = idx >> 2;
        int seg = idx & 3;
        const unsigned short* gp = G + (row0 + row) * (size_t)CDIM + k0 + seg * 8;
        unsigned short* lp = &S[row][seg * 8];
#if USE_ASYNC_LDS
        async_copy16(gp, lp);
#else
        const uint4 v = *(const uint4*)gp;
        unsigned int* d = (unsigned int*)lp;
        d[0] = v.x; d[1] = v.y; d[2] = v.z; d[3] = v.w;
#endif
    }
}

static __device__ __forceinline__ void wait_stage() {
#if USE_ASYNC_LDS
    __builtin_amdgcn_s_wait_asynccnt(0);
#endif
    __syncthreads();
}

// ---------------------------------------------------------------------------
// Weight pre-conversion fp32 -> bf16 (done once; makes GEMM B-tiles raw copies)
// ---------------------------------------------------------------------------
__global__ __launch_bounds__(256) void cvt_bf16_kernel(
    const float* __restrict__ src, unsigned short* __restrict__ dst, int n)
{
    int i = (blockIdx.x * 256 + threadIdx.x) * 4;
    if (i < n) {
        float4 v = *(const float4*)&src[i];
        dst[i + 0] = f32_to_bf16(v.x);
        dst[i + 1] = f32_to_bf16(v.y);
        dst[i + 2] = f32_to_bf16(v.z);
        dst[i + 3] = f32_to_bf16(v.w);
    }
}

// ---------------------------------------------------------------------------
// GEMM 1: out = mix(x, last_x) @ W^T   (mode 0: fp32 out, mode 1: sigmoid->bf16)
// A: fp32 x with per-channel time-mix (VALU staging). B: bf16 weights (async).
// ---------------------------------------------------------------------------
__global__ __launch_bounds__(256) void mix_gemm_kernel(
    const float* __restrict__ X,            // (M, C) fp32
    const float* __restrict__ LX,           // (N, 1, C) broadcast over T
    const unsigned short* __restrict__ W16, // (Cout, C) bf16
    const float* __restrict__ TM,           // (C,) time-mix
    float* __restrict__ outF,
    unsigned short* __restrict__ outB,
    int mode)
{
    __shared__ unsigned short As[BM][LDK];
    __shared__ unsigned short Bs[BN][LDK];

    const int m0   = blockIdx.x * BM;
    const int n0   = blockIdx.y * BN;
    const int nIdx = m0 / TLEN;                 // BM divides TLEN
    const float* lx = LX + (size_t)nIdx * CDIM;

    const int tid    = threadIdx.x;
    const int lane   = tid & 31;
    const int wave   = tid >> 5;                // 8 waves
    const int wave_m = wave & 3;                // 4 along M (32 rows)
    const int wave_n = wave >> 2;               // 2 along N (64 cols)
    const int half   = lane >> 4;
    const int l16    = lane & 15;

    v8f acc[2][4] = {};

    for (int k0 = 0; k0 < CDIM; k0 += KC) {
        // B tile: raw bf16 copy (async global->LDS when available)
        stage_tile_bf16(W16, (size_t)n0, k0, Bs, tid);

        // A tile: fp32 load + time-mix + bf16 convert (overlaps async B)
        #pragma unroll
        for (int it = 0; it < 4; ++it) {
            int idx  = tid + it * 256;          // over 128 rows x 8 float4
            int row  = idx >> 3;
            int c4   = idx & 7;
            int c    = k0 + c4 * 4;
            const float4 xv = *(const float4*)&X[(size_t)(m0 + row) * CDIM + c];
            const float4 tm = *(const float4*)&TM[c];
            const float4 lv = *(const float4*)&lx[c];
            unsigned int* dst = (unsigned int*)&As[row][0];
            unsigned short p0 = f32_to_bf16(xv.x * tm.x + lv.x * (1.f - tm.x));
            unsigned short p1 = f32_to_bf16(xv.y * tm.y + lv.y * (1.f - tm.y));
            unsigned short p2 = f32_to_bf16(xv.z * tm.z + lv.z * (1.f - tm.z));
            unsigned short p3 = f32_to_bf16(xv.w * tm.w + lv.w * (1.f - tm.w));
            dst[c4 * 2 + 0] = (unsigned int)p0 | ((unsigned int)p1 << 16);
            dst[c4 * 2 + 1] = (unsigned int)p2 | ((unsigned int)p3 << 16);
        }
        if (k0 + KC < CDIM)
            __builtin_prefetch(&X[(size_t)(m0 + (tid >> 3)) * CDIM + k0 + KC], 0, 0);

        wait_stage();

        FragU af[2], bf[4];
        for (int i = 0; i < 2; ++i) {
            int row = wave_m * 32 + i * 16 + l16;
            const unsigned int* ar = (const unsigned int*)&As[row][0];
            #pragma unroll
            for (int pr = 0; pr < 8; ++pr) af[i].u[pr] = ar[frag_kpair(pr, half)];
        }
        for (int j = 0; j < 4; ++j) {
            int col = wave_n * 64 + j * 16 + l16;
            const unsigned int* br = (const unsigned int*)&Bs[col][0];
            #pragma unroll
            for (int pr = 0; pr < 8; ++pr) bf[j].u[pr] = br[frag_kpair(pr, half)];
        }
        #pragma unroll
        for (int i = 0; i < 2; ++i)
            #pragma unroll
            for (int j = 0; j < 4; ++j)
                acc[i][j] = __builtin_amdgcn_wmma_f32_16x16x32_bf16(
                    false, af[i].v, false, bf[j].v,
                    (short)0, acc[i][j], false, false);
        __syncthreads();
    }

    // Store: C/D layout VGPR r -> M = r + 8*half, N = lane%16
    for (int i = 0; i < 2; ++i) {
        for (int j = 0; j < 4; ++j) {
            int gm0 = m0 + wave_m * 32 + i * 16 + 8 * half;
            int gc  = n0 + wave_n * 64 + j * 16 + l16;
            #pragma unroll
            for (int r = 0; r < 8; ++r) {
                float v = acc[i][j][r];
                size_t o = (size_t)(gm0 + r) * CDIM + gc;
                if (mode == 0) outF[o] = v;
                else           outB[o] = f32_to_bf16(1.0f / (1.0f + __expf(-v)));
            }
        }
    }
}

// ---------------------------------------------------------------------------
// wkv numerically-stabilized scan, fused with * sigmoid(r), bf16 output
// ---------------------------------------------------------------------------
__global__ __launch_bounds__(256) void wkv_scan_kernel(
    const float* __restrict__ K, const float* __restrict__ V,
    const unsigned short* __restrict__ SR,
    const float* __restrict__ time_decay, const float* __restrict__ time_first,
    const float* __restrict__ state,      // (N, 3, 1, C)
    unsigned short* __restrict__ Y)
{
    const int id = blockIdx.x * blockDim.x + threadIdx.x;  // N*C threads
    const int n  = id / CDIM;
    const int c  = id % CDIM;

    float a = state[(size_t)n * 3 * CDIM + 0 * CDIM + c];
    float b = state[(size_t)n * 3 * CDIM + 1 * CDIM + c];
    float e = state[(size_t)n * 3 * CDIM + 2 * CDIM + c];
    const float w = __expf(time_decay[c]);
    const float u = time_first[c];

    size_t idx = (size_t)n * TLEN * CDIM + c;
    for (int t = 0; t < TLEN; ++t, idx += CDIM) {
        float kt  = K[idx];
        float vt  = V[idx];
        float ukt = u + kt;
        float tau = fmaxf(ukt, e);
        float e1  = __expf(e - tau);
        float e2  = __expf(ukt - tau);
        float wkv = (e1 * a + e2 * vt) / (e1 * b + e2);
        Y[idx] = f32_to_bf16(wkv * bf16_to_f32(SR[idx]));
        float w_eps = e - w;
        float e_n   = fmaxf(w_eps, kt);
        float e1n   = __expf(w_eps - e_n);
        float e2n   = __expf(kt - e_n);
        a = e1n * a + e2n * vt;
        b = e1n * b + e2n;
        e = e_n;
    }
}

// ---------------------------------------------------------------------------
// GEMM 2: out = Y @ Wo^T   (Y bf16, Wo bf16; both tiles raw async copies)
// ---------------------------------------------------------------------------
__global__ __launch_bounds__(256) void out_gemm_kernel(
    const unsigned short* __restrict__ Y,    // (M, C) bf16
    const unsigned short* __restrict__ Wo16, // (NCLS, C) bf16
    float* __restrict__ OUT)                 // (M, NCLS) fp32
{
    __shared__ unsigned short As[BM][LDK];
    __shared__ unsigned short Bs[BN][LDK];

    const int m0 = blockIdx.x * BM;
    const int n0 = blockIdx.y * BN;

    const int tid    = threadIdx.x;
    const int lane   = tid & 31;
    const int wave   = tid >> 5;
    const int wave_m = wave & 3;
    const int wave_n = wave >> 2;
    const int half   = lane >> 4;
    const int l16    = lane & 15;

    v8f acc[2][4] = {};

    for (int k0 = 0; k0 < CDIM; k0 += KC) {
        stage_tile_bf16(Y,    (size_t)m0, k0, As, tid);
        stage_tile_bf16(Wo16, (size_t)n0, k0, Bs, tid);
        wait_stage();

        FragU af[2], bf[4];
        for (int i = 0; i < 2; ++i) {
            int row = wave_m * 32 + i * 16 + l16;
            const unsigned int* ar = (const unsigned int*)&As[row][0];
            #pragma unroll
            for (int pr = 0; pr < 8; ++pr) af[i].u[pr] = ar[frag_kpair(pr, half)];
        }
        for (int j = 0; j < 4; ++j) {
            int col = wave_n * 64 + j * 16 + l16;
            const unsigned int* br = (const unsigned int*)&Bs[col][0];
            #pragma unroll
            for (int pr = 0; pr < 8; ++pr) bf[j].u[pr] = br[frag_kpair(pr, half)];
        }
        #pragma unroll
        for (int i = 0; i < 2; ++i)
            #pragma unroll
            for (int j = 0; j < 4; ++j)
                acc[i][j] = __builtin_amdgcn_wmma_f32_16x16x32_bf16(
                    false, af[i].v, false, bf[j].v,
                    (short)0, acc[i][j], false, false);
        __syncthreads();
    }

    for (int i = 0; i < 2; ++i) {
        for (int j = 0; j < 4; ++j) {
            int gm0 = m0 + wave_m * 32 + i * 16 + 8 * half;
            int gc  = n0 + wave_n * 64 + j * 16 + l16;
            #pragma unroll
            for (int r = 0; r < 8; ++r)
                OUT[(size_t)(gm0 + r) * NCLS + gc] = acc[i][j][r];
        }
    }
}

// ---------------------------------------------------------------------------
extern "C" void kernel_launch(void* const* d_in, const int* in_sizes, int n_in,
                              void* d_out, int out_size, void* d_ws, size_t ws_size,
                              hipStream_t stream) {
    const float* x      = (const float*)d_in[0];
    const float* Wk     = (const float*)d_in[2];
    const float* Wv     = (const float*)d_in[3];
    const float* Wr     = (const float*)d_in[4];
    const float* Wo     = (const float*)d_in[5];
    const float* tmk    = (const float*)d_in[6];
    const float* tmv    = (const float*)d_in[7];
    const float* tmr    = (const float*)d_in[8];
    const float* tdec   = (const float*)d_in[9];
    const float* tfirst = (const float*)d_in[10];
    const float* lastx  = (const float*)d_in[11];
    const float* state  = (const float*)d_in[12];

    // workspace layout
    const size_t szF = (size_t)MROWS * CDIM * sizeof(float);          // 64 MB
    const size_t szB = (size_t)MROWS * CDIM * sizeof(unsigned short); // 32 MB
    const size_t szW = (size_t)NCLS * CDIM * sizeof(unsigned short);  //  2 MB
    char* ws = (char*)d_ws;
    float*          kbuf  = (float*)(ws);
    float*          vbuf  = (float*)(ws + szF);
    unsigned short* srbuf = (unsigned short*)(ws + 2 * szF);
    unsigned short* ybuf  = (unsigned short*)(ws + 2 * szF + szB);
    unsigned short* wk16  = (unsigned short*)(ws + 2 * szF + 2 * szB);
    unsigned short* wv16  = (unsigned short*)(ws + 2 * szF + 2 * szB + szW);
    unsigned short* wr16  = (unsigned short*)(ws + 2 * szF + 2 * szB + 2 * szW);
    unsigned short* wo16  = (unsigned short*)(ws + 2 * szF + 2 * szB + 3 * szW);

    dim3 block(256);
    const int nW = NCLS * CDIM;
    dim3 gridC(nW / (256 * 4));
    cvt_bf16_kernel<<<gridC, block, 0, stream>>>(Wk, wk16, nW);
    cvt_bf16_kernel<<<gridC, block, 0, stream>>>(Wv, wv16, nW);
    cvt_bf16_kernel<<<gridC, block, 0, stream>>>(Wr, wr16, nW);
    cvt_bf16_kernel<<<gridC, block, 0, stream>>>(Wo, wo16, nW);

    dim3 gridG(MROWS / BM, NCLS / BN);   // 128 x 8
    mix_gemm_kernel<<<gridG, block, 0, stream>>>(x, lastx, wk16, tmk, kbuf, nullptr, 0);
    mix_gemm_kernel<<<gridG, block, 0, stream>>>(x, lastx, wv16, tmv, vbuf, nullptr, 0);
    mix_gemm_kernel<<<gridG, block, 0, stream>>>(x, lastx, wr16, tmr, nullptr, srbuf, 1);

    wkv_scan_kernel<<<(NB * CDIM) / 256, block, 0, stream>>>(
        kbuf, vbuf, srbuf, tdec, tfirst, state, ybuf);

    out_gemm_kernel<<<gridG, block, 0, stream>>>(ybuf, wo16, (float*)d_out);
}
